// PointFlowModuleWithMaxAvgpool_8040178778560
// MI455X (gfx1250) — matrix operand order repaired and checked by reading, not compiled, not run on hardware
//
#include <hip/hip_runtime.h>
#include <hip/hip_bf16.h>
#include <math.h>

// ---------------- problem constants ----------------
#define NB   2
#define CP   256
#define DIM  64
#define HH   128
#define WH   128
#define HWH  (HH*WH)        // 16384
#define HL   64
#define WL   64
#define HWL  (HL*WL)        // 4096
#define POOL 8
#define PEDGE 64
#define BN_EPS 1e-5f

typedef __attribute__((ext_vector_type(16))) __bf16 bf16x16;
typedef __attribute__((ext_vector_type(8)))  float  floatx8;
typedef int v4i __attribute__((ext_vector_type(4)));

// gfx1250 async global->LDS copy path (ASYNCcnt); guarded so the file always
// compiles. Fallback is a plain vectorized LDS store copy.
#if defined(__AMDGCN__) && \
    __has_builtin(__builtin_amdgcn_global_load_async_to_lds_b128) && \
    __has_builtin(__builtin_amdgcn_s_wait_asynccnt)
#define USE_ASYNC_LDS 1
typedef __attribute__((address_space(1))) v4i gv4i;  // global int4
typedef __attribute__((address_space(3))) v4i lv4i;  // LDS int4
#else
#define USE_ASYNC_LDS 0
#endif

// ======================================================================
// K0a: f32 -> bf16 conversion
// ======================================================================
__global__ void convert_bf16_kernel(const float* __restrict__ src,
                                    __bf16* __restrict__ dst, int n) {
    int i = blockIdx.x * blockDim.x + threadIdx.x;
    if (i < n) dst[i] = (__bf16)src[i];
}

// ======================================================================
// K0b: w_edge1 OIHW f32 -> bf16 transposed to [tap][kchunk][co][k_local]
// so each (tap,kchunk) B-slice is one contiguous 16KB block.
// out[((t*8+kc)*256+co)*32+kl] = in[(co*256 + kc*32 + kl)*9 + t]
// ======================================================================
__global__ void transpose_we1_kernel(const float* __restrict__ src,
                                     __bf16* __restrict__ dst) {
    int o = blockIdx.x * blockDim.x + threadIdx.x;
    if (o >= CP * CP * 9) return;
    int kl = o & 31;
    int co = (o >> 5) & 255;
    int kc = (o >> 13) & 7;
    int t  = o >> 16;
    dst[o] = (__bf16)src[(co * CP + kc * 32 + kl) * 9 + t];
}

// ======================================================================
// K1: 1x1 conv as WMMA GEMM.  out[nb,d,pix] = sum_c W[d,c]*src[nb,c,pix] + b[d]
// block: 128 thr = 4 waves; 16 pixels per block; wave w -> out ch [16w,16w+16)
// ======================================================================
__global__ __launch_bounds__(128)
void conv1x1_wmma_kernel(const float* __restrict__ src,
                         const __bf16* __restrict__ wb,   // [DIM][CP] bf16
                         const float* __restrict__ bias,  // [DIM]
                         float* __restrict__ dst, int HW) {
    __shared__ __align__(16) __bf16 aT[16 * CP];
    const int tid = threadIdx.x;
    const int gpixBase = blockIdx.x * 16;
    const int nb = gpixBase / HW;
    const int pixBase = gpixBase % HW;

    // stage A tile (16 pixels x 256 K) as bf16
    for (int e = tid; e < 16 * CP; e += 128) {
        int p = e >> 8, ci = e & 255;
        aT[p * CP + ci] = (__bf16)src[((size_t)(nb * CP + ci)) * HW + pixBase + p];
    }
    __syncthreads();

    const int wave = tid >> 5, lane = tid & 31;
    const int ncol = lane & 15, khalf = lane >> 4;
    const int m = lane & 15;
    const int coBase = wave * 16;

    floatx8 acc = {0.f,0.f,0.f,0.f,0.f,0.f,0.f,0.f};
    #pragma unroll
    for (int kc = 0; kc < 8; ++kc) {
        const int base = kc * 32;
        bf16x16 a, b;
        #pragma unroll
        for (int i = 0; i < 16; ++i) {
            a[i]     = aT[m * CP + base + khalf * 16 + i];
        }
        const int kb = base + khalf * 16;
        #pragma unroll
        for (int i = 0; i < 16; ++i)
            b[i] = wb[(coBase + ncol) * CP + kb + i];
        acc = __builtin_amdgcn_wmma_f32_16x16x32_bf16(false, a, false, b,
                                                      (short)0, acc, false, false);
    }
    const int co = coBase + ncol;
    const float bv = bias[co];
    const int mOff = khalf * 8;
    #pragma unroll
    for (int i = 0; i < 8; ++i) {
        int pix = pixBase + i + mOff;
        dst[((size_t)(nb * DIM + co)) * HW + pix] = acc[i] + bv;
    }
}

// ======================================================================
// K2a: bilinear resize, align_corners=True.  src (NC,h,w) -> dst (NC,oh,ow)
// ======================================================================
__global__ void resize_ac_kernel(const float* __restrict__ src, float* __restrict__ dst,
                                 int nc, int h, int w, int oh, int ow) {
    int idx = blockIdx.x * blockDim.x + threadIdx.x;
    int total = nc * oh * ow;
    if (idx >= total) return;
    int x = idx % ow, y = (idx / ow) % oh, c = idx / (ow * oh);
    float py = (oh > 1) ? y * (float)(h - 1) / (float)(oh - 1) : 0.f;
    float px = (ow > 1) ? x * (float)(w - 1) / (float)(ow - 1) : 0.f;
    int y0 = (int)floorf(py), x0 = (int)floorf(px);
    int y1 = min(y0 + 1, h - 1), x1 = min(x0 + 1, w - 1);
    float wy = py - y0, wx = px - x0;
    const float* s = src + (size_t)c * h * w;
    float r0 = s[y0 * w + x0] * (1.f - wx) + s[y0 * w + x1] * wx;
    float r1 = s[y1 * w + x0] * (1.f - wx) + s[y1 * w + x1] * wx;
    dst[idx] = r0 * (1.f - wy) + r1 * wy;
}

// ======================================================================
// K2b: matcher 3x3 conv over concat(xh_embed, low_up) (128 ch) -> sigmoid
// ======================================================================
__global__ void match_conv_kernel(const float* __restrict__ xhE,
                                  const float* __restrict__ lowUp,
                                  const float* __restrict__ wM,  // [1][128][3][3]
                                  const float* __restrict__ bM,
                                  float* __restrict__ cert) {
    int idx = blockIdx.x * blockDim.x + threadIdx.x;
    if (idx >= NB * HWH) return;
    int n = idx / HWH, pix = idx % HWH;
    int y = pix >> 7, x = pix & 127;
    float s = bM[0];
    for (int ci = 0; ci < 2 * DIM; ++ci) {
        const float* img = (ci < DIM) ? (xhE + (size_t)(n * DIM + ci) * HWH)
                                      : (lowUp + (size_t)(n * DIM + (ci - DIM)) * HWH);
        #pragma unroll
        for (int dy = -1; dy <= 1; ++dy)
            #pragma unroll
            for (int dx = -1; dx <= 1; ++dx) {
                int sy = y + dy, sx = x + dx;
                if ((unsigned)sy < (unsigned)HH && (unsigned)sx < (unsigned)WH)
                    s += img[sy * WH + sx] * wM[ci * 9 + (dy + 1) * 3 + (dx + 1)];
            }
    }
    cert[idx] = 1.f / (1.f + expf(-s));
}

// ======================================================================
// K3: 8x8 adaptive avg + max-with-index over certainty
// ======================================================================
__global__ void pool_kernel(const float* __restrict__ cert,
                            float* __restrict__ avgP, float* __restrict__ maxV,
                            int* __restrict__ maxI) {
    int idx = blockIdx.x * blockDim.x + threadIdx.x;
    if (idx >= NB * POOL * POOL) return;
    int n = idx / (POOL * POOL), cell = idx % (POOL * POOL);
    int py = cell / POOL, px = cell % POOL;
    const int kh = HH / POOL, kw = WH / POOL;  // 16 x 16
    const float* c = cert + (size_t)n * HWH;
    float sum = 0.f, best = -INFINITY;
    int bi = 0;
    for (int r = 0; r < kh; ++r)
        for (int q = 0; q < kw; ++q) {
            int yy = py * kh + r, xx = px * kw + q;
            float v = c[yy * WH + xx];
            sum += v;
            if (v > best) { best = v; bi = yy * WH + xx; }
        }
    avgP[idx] = sum / (float)(kh * kw);
    maxV[idx] = best;
    maxI[idx] = bi;
}

// align-corners bilinear sample of an 8x8 pool map at high-res pixel (y,x)
__device__ __forceinline__ float pool_up8(const float* p, int y, int x) {
    float py = y * (7.0f / 127.0f), px = x * (7.0f / 127.0f);
    int y0 = (int)py, x0 = (int)px;
    int y1 = min(y0 + 1, 7), x1 = min(x0 + 1, 7);
    float wy = py - y0, wx = px - x0;
    float r0 = p[y0 * 8 + x0] * (1.f - wx) + p[y0 * 8 + x1] * wx;
    float r1 = p[y1 * 8 + x0] * (1.f - wx) + p[y1 * 8 + x1] * wx;
    return r0 * (1.f - wy) + r1 * wy;
}

// ======================================================================
// K4: edge1 3x3 conv 256->256 via WMMA implicit GEMM.
//  - input scale (1-avg_up) applied on the fly while staging A tile (bf16)
//  - per (tap,kchunk) weight slice (contiguous 16KB, pre-transposed) staged
//    into LDS with GLOBAL_LOAD_ASYNC_TO_LDS_B128 when available
//  - BN+ReLU fused at store; output bf16
// block: 512 thr = 16 waves; 16 pixels; wave w -> out ch [16w,16w+16)
// ======================================================================
__global__ __launch_bounds__(512)
void conv3x3_edge1_wmma_kernel(const float* __restrict__ xh,
                               const __bf16* __restrict__ we1T, // [9][8][256][32] bf16
                               const float* __restrict__ avgP,
                               const float* __restrict__ bnG,
                               const float* __restrict__ bnB,
                               const float* __restrict__ bnM,
                               const float* __restrict__ bnV,
                               __bf16* __restrict__ hbuf) {
    __shared__ __align__(16) __bf16 aT[16 * CP];     //  8 KB
    __shared__ __align__(16) __bf16 bT[32 * CP];     // 16 KB: [co][k_local]
    __shared__ float sPool[64];
    __shared__ float sScale[16];
    const int tid = threadIdx.x;
    const int gpixBase = blockIdx.x * 16;
    const int nb = gpixBase / HWH;
    const int pixBase = gpixBase % HWH;
    if (tid < 64) sPool[tid] = avgP[nb * 64 + tid];

    const int wave = tid >> 5, lane = tid & 31;
    const int ncol = lane & 15, khalf = lane >> 4;
    const int m = lane & 15;
    const int coBase = wave * 16;

    floatx8 acc = {0.f,0.f,0.f,0.f,0.f,0.f,0.f,0.f};

    for (int t = 0; t < 9; ++t) {
        const int dy = t / 3 - 1, dx = t % 3 - 1;
        __syncthreads();   // prior tap's aT/sScale reads done (also covers sPool)
        if (tid < 16) {
            int gpix = pixBase + tid;
            int y = (gpix >> 7) + dy, x = (gpix & 127) + dx;
            float s = 0.f;
            if ((unsigned)y < (unsigned)HH && (unsigned)x < (unsigned)WH)
                s = 1.0f - pool_up8(sPool, y, x);
            sScale[tid] = s;
        }
        __syncthreads();
        // stage scaled, zero-padded A tap tile (transform: f32 -> scaled bf16)
        for (int e = tid; e < 16 * CP; e += 512) {
            int p = e >> 8, ci = e & 255;
            int gpix = pixBase + p;
            int y = (gpix >> 7) + dy, x = (gpix & 127) + dx;
            float v = 0.f;
            if ((unsigned)y < (unsigned)HH && (unsigned)x < (unsigned)WH)
                v = xh[(((size_t)(nb * CP + ci)) << 14) + (y << 7) + x] * sScale[p];
            aT[p * CP + ci] = (__bf16)v;
        }
        for (int kc = 0; kc < 8; ++kc) {
            __syncthreads();  // aT staged (kc==0) / prior bT reads done
            // ---- stage 16KB contiguous B slice into LDS ----
            {
                const char* gsrc =
                    (const char*)(we1T + ((size_t)(t * 8 + kc)) * 32 * CP);
                char* lbase = (char*)bT;
#if USE_ASYNC_LDS
                __builtin_amdgcn_global_load_async_to_lds_b128(
                    (gv4i*)(gsrc + tid * 16),
                    (lv4i*)(lbase + tid * 16), 0, 0);
                __builtin_amdgcn_global_load_async_to_lds_b128(
                    (gv4i*)(gsrc + 8192 + tid * 16),
                    (lv4i*)(lbase + 8192 + tid * 16), 0, 0);
                __builtin_amdgcn_s_wait_asynccnt(0);
#else
                const uint4* gs = (const uint4*)gsrc;
                uint4* ld = (uint4*)lbase;
                ld[tid]       = gs[tid];
                ld[tid + 512] = gs[tid + 512];
#endif
            }
            __syncthreads();  // bT visible to all waves
            const int base = kc * 32;
            bf16x16 a, b;
            #pragma unroll
            for (int i = 0; i < 16; ++i) {
                a[i]     = aT[m * CP + base + khalf * 16 + i];
            }
            const __bf16* bp = bT + (coBase + ncol) * 32 + khalf * 16;
            #pragma unroll
            for (int i = 0; i < 16; ++i)
                b[i] = bp[i];
            acc = __builtin_amdgcn_wmma_f32_16x16x32_bf16(false, a, false, b,
                                                          (short)0, acc, false, false);
        }
    }
    // fused BN + ReLU, store bf16
    const int co = coBase + ncol;
    const float inv = rsqrtf(bnV[co] + BN_EPS);
    const float sc = inv * bnG[co];
    const float mean = bnM[co], beta = bnB[co];
    const int mOff = khalf * 8;
    #pragma unroll
    for (int i = 0; i < 8; ++i) {
        float h = (acc[i] - mean) * sc + beta;
        h = fmaxf(h, 0.f);
        hbuf[(((size_t)(nb * CP + co)) << 14) + pixBase + i + mOff] = (__bf16)h;
    }
}

// ======================================================================
// K5: edge2 3x3 conv 256->1 (no bias); writes edge_pred (d_out tail) + scratch
// ======================================================================
__global__ void edge2_conv_kernel(const __bf16* __restrict__ hbuf,
                                  const float* __restrict__ w2, // [1][CP][3][3]
                                  float* __restrict__ edgeOut,
                                  float* __restrict__ scratch) {
    int idx = blockIdx.x * blockDim.x + threadIdx.x;
    if (idx >= NB * HWH) return;
    int n = idx / HWH, pix = idx % HWH;
    int y = pix >> 7, x = pix & 127;
    float s = 0.f;
    for (int co = 0; co < CP; ++co) {
        const __bf16* hp = hbuf + (((size_t)(n * CP + co)) << 14);
        #pragma unroll
        for (int dy = -1; dy <= 1; ++dy)
            #pragma unroll
            for (int dx = -1; dx <= 1; ++dx) {
                int sy = y + dy, sx = x + dx;
                if ((unsigned)sy < (unsigned)HH && (unsigned)sx < (unsigned)WH)
                    s += (float)hp[sy * WH + sx] * w2[co * 9 + (dy + 1) * 3 + (dx + 1)];
            }
    }
    edgeOut[idx] = s;
    scratch[idx] = s;
}

// ======================================================================
// K6: top-64 per batch via 64 rounds of block-wide argmax (destroys scratch)
// ======================================================================
__global__ __launch_bounds__(256)
void topk_kernel(float* __restrict__ scratch, int* __restrict__ topkI,
                 int* __restrict__ lowEdgeI) {
    __shared__ float sv[256];
    __shared__ int   si[256];
    const int n = blockIdx.x, t = threadIdx.x;
    float* s = scratch + (size_t)n * HWH;
    for (int it = 0; it < PEDGE; ++it) {
        float best = -INFINITY; int bi = 0;
        for (int e = t; e < HWH; e += 256) {
            float v = s[e];
            if (v > best) { best = v; bi = e; }
        }
        sv[t] = best; si[t] = bi;
        __syncthreads();
        for (int st = 128; st > 0; st >>= 1) {
            if (t < st && sv[t + st] > sv[t]) { sv[t] = sv[t + st]; si[t] = si[t + st]; }
            __syncthreads();
        }
        if (t == 0) {
            int b = si[0];
            topkI[n * PEDGE + it] = b;
            s[b] = -INFINITY;
            int ix = b & 127, iy = b >> 7;
            lowEdgeI[n * PEDGE + it] = (int)(ix * 0.5f + iy * 0.5f * (float)WL);
        }
        __syncthreads();
    }
}

// bilinear with zeros padding, grid_sample convention (pixel coords)
__device__ __forceinline__ float bilin_zero(const float* img, int H, int W,
                                            float px, float py) {
    float x0f = floorf(px), y0f = floorf(py);
    int x0 = (int)x0f, y0 = (int)y0f;
    float wx = px - x0f, wy = py - y0f;
    float acc = 0.f;
    #pragma unroll
    for (int dy = 0; dy < 2; ++dy)
        #pragma unroll
        for (int dx = 0; dx < 2; ++dx) {
            int yi = y0 + dy, xi = x0 + dx;
            float wgt = (dy ? wy : 1.f - wy) * (dx ? wx : 1.f - wx);
            if ((unsigned)yi < (unsigned)H && (unsigned)xi < (unsigned)W)
                acc += img[yi * W + xi] * wgt;
        }
    return acc;
}

// ======================================================================
// K7: gather edge points: high_e (exact pixel) / low_e (true bilinear)
// ======================================================================
__global__ void gather_edge_kernel(const float* __restrict__ xh,
                                   const float* __restrict__ xl,
                                   const int* __restrict__ topkI,
                                   float* __restrict__ highE,
                                   float* __restrict__ lowE) {
    int idx = blockIdx.x * blockDim.x + threadIdx.x;
    if (idx >= NB * CP * PEDGE) return;
    int n = idx / (CP * PEDGE);
    int c = (idx / PEDGE) % CP;
    int p = idx % PEDGE;
    int g = topkI[n * PEDGE + p];
    int ix = g & 127, iy = g >> 7;
    float cx = (0.5f + (float)ix) / (float)WH;
    float cy = (0.5f + (float)iy) / (float)HH;
    const float* ih = xh + (((size_t)(n * CP + c)) << 14);
    const float* il = xl + (((size_t)(n * CP + c)) << 12);
    highE[idx] = bilin_zero(ih, HH, WH, cx * WH - 0.5f, cy * HH - 0.5f);
    lowE[idx]  = bilin_zero(il, HL, WL, cx * WL - 0.5f, cy * HL - 0.5f);
}

// ======================================================================
// K8: attention fusion: out[c][q] = sum_p softmax_p(high·low_q)[p]*high[c][p] + low[c][q]
// one block per (n,q), 64 threads (p)
// ======================================================================
__global__ __launch_bounds__(64)
void attn_fuse_kernel(const float* __restrict__ high, const float* __restrict__ low,
                      float* __restrict__ out) {
    __shared__ float lowq[CP];
    __shared__ float aff[PEDGE];
    __shared__ float red[PEDGE];
    const int n = blockIdx.x / PEDGE, q = blockIdx.x % PEDGE, t = threadIdx.x;
    for (int c = t; c < CP; c += 64)
        lowq[c] = low[((size_t)(n * CP + c)) * PEDGE + q];
    __syncthreads();
    float d = 0.f;
    for (int c = 0; c < CP; ++c)
        d += high[((size_t)(n * CP + c)) * PEDGE + t] * lowq[c];
    red[t] = d; __syncthreads();
    for (int s = 32; s > 0; s >>= 1) {
        if (t < s) red[t] = fmaxf(red[t], red[t + s]);
        __syncthreads();
    }
    float mx = red[0]; __syncthreads();
    float e = expf(d - mx);
    red[t] = e; __syncthreads();
    for (int s = 32; s > 0; s >>= 1) {
        if (t < s) red[t] += red[t + s];
        __syncthreads();
    }
    aff[t] = e / red[0];
    __syncthreads();
    for (int c = t; c < CP; c += 64) {
        float s = 0.f;
        const float* hp = high + ((size_t)(n * CP + c)) * PEDGE;
        for (int p = 0; p < PEDGE; ++p) s += aff[p] * hp[p];
        out[((size_t)(n * CP + c)) * PEDGE + q] = s + lowq[c];
    }
}

// ======================================================================
// K9: gather max-branch features with on-the-fly (1+max_up) scale
// ======================================================================
__global__ void gather_max_kernel(const float* __restrict__ xh,
                                  const float* __restrict__ xl,
                                  const int* __restrict__ maxI,
                                  const float* __restrict__ maxV,
                                  float* __restrict__ highF,
                                  float* __restrict__ lowF,
                                  int* __restrict__ lowFeatI) {
    int idx = blockIdx.x * blockDim.x + threadIdx.x;
    if (idx >= NB * CP * PEDGE) return;
    int n = idx / (CP * PEDGE);
    int c = (idx / PEDGE) % CP;
    int p = idx % PEDGE;
    int flat = maxI[n * PEDGE + p];
    int y = flat >> 7, x = flat & 127;
    float mup = pool_up8(maxV + n * 64, y, x);
    highF[idx] = xh[(((size_t)(n * CP + c)) << 14) + flat] * (1.f + mup);
    int li = (int)(x * 0.5f + y * 0.5f * (float)WL);
    lowF[idx] = xl[(((size_t)(n * CP + c)) << 12) + li];
    if (c == 0) lowFeatI[n * PEDGE + p] = li;
}

// ======================================================================
// K10-12: assemble final output
// ======================================================================
__global__ void copy_low_kernel(const float* __restrict__ xl, float* __restrict__ out) {
    int i = blockIdx.x * blockDim.x + threadIdx.x;
    if (i < NB * CP * HWL) out[i] = xl[i];
}

__global__ void scatter_kernel(const float* __restrict__ fused,
                               const int* __restrict__ lidx,
                               float* __restrict__ out) {
    int idx = blockIdx.x * blockDim.x + threadIdx.x;
    if (idx >= NB * CP * PEDGE) return;
    int n = idx / (CP * PEDGE);
    int c = (idx / PEDGE) % CP;
    int p = idx % PEDGE;
    int li = lidx[n * PEDGE + p];
    out[(((size_t)(n * CP + c)) << 12) + li] = fused[idx];
}

// ======================================================================
// launch
// ======================================================================
extern "C" void kernel_launch(void* const* d_in, const int* in_sizes, int n_in,
                              void* d_out, int out_size, void* d_ws, size_t ws_size,
                              hipStream_t stream) {
    const float* x_high  = (const float*)d_in[0];
    const float* x_low   = (const float*)d_in[1];
    const float* w_dh    = (const float*)d_in[2];
    const float* b_dh    = (const float*)d_in[3];
    const float* w_dl    = (const float*)d_in[4];
    const float* b_dl    = (const float*)d_in[5];
    const float* w_match = (const float*)d_in[6];
    const float* b_match = (const float*)d_in[7];
    const float* w_e1    = (const float*)d_in[8];
    const float* bn_g    = (const float*)d_in[9];
    const float* bn_b    = (const float*)d_in[10];
    const float* bn_m    = (const float*)d_in[11];
    const float* bn_v    = (const float*)d_in[12];
    const float* w_e2    = (const float*)d_in[13];

    float* outFinal = (float*)d_out;                         // (2,256,64,64)
    float* outEdge  = (float*)d_out + (size_t)NB * CP * HWL; // (2,1,128,128)

    // ---- workspace carve-out ----
    char* ws = (char*)d_ws;
    size_t off = 0;
    auto carve = [&](size_t bytes) -> char* {
        char* p = ws + off;
        off += (bytes + 255) & ~(size_t)255;
        return p;
    };
    float* xhE    = (float*)carve((size_t)NB * DIM * HWH * 4);
    float* xlE    = (float*)carve((size_t)NB * DIM * HWL * 4);
    float* lowUp  = (float*)carve((size_t)NB * DIM * HWH * 4);
    float* cert   = (float*)carve((size_t)NB * HWH * 4);
    float* avgP   = (float*)carve((size_t)NB * 64 * 4);
    float* maxV   = (float*)carve((size_t)NB * 64 * 4);
    int*   maxI   = (int*)  carve((size_t)NB * 64 * 4);
    __bf16* wdhB  = (__bf16*)carve((size_t)DIM * CP * 2);
    __bf16* wdlB  = (__bf16*)carve((size_t)DIM * CP * 2);
    __bf16* we1T  = (__bf16*)carve((size_t)CP * CP * 9 * 2);
    __bf16* hbuf  = (__bf16*)carve((size_t)NB * CP * HWH * 2);
    float* scratch= (float*)carve((size_t)NB * HWH * 4);
    int*   topkI  = (int*)  carve((size_t)NB * PEDGE * 4);
    int*   lowEI  = (int*)  carve((size_t)NB * PEDGE * 4);
    int*   lowFI  = (int*)  carve((size_t)NB * PEDGE * 4);
    float* highE  = (float*)carve((size_t)NB * CP * PEDGE * 4);
    float* lowE   = (float*)carve((size_t)NB * CP * PEDGE * 4);
    float* fusE   = (float*)carve((size_t)NB * CP * PEDGE * 4);
    float* highF  = (float*)carve((size_t)NB * CP * PEDGE * 4);
    float* lowF   = (float*)carve((size_t)NB * CP * PEDGE * 4);
    float* fusF   = (float*)carve((size_t)NB * CP * PEDGE * 4);
    (void)ws_size; (void)n_in; (void)in_sizes; (void)out_size;

    // K0: weight conversions / transpose
    {
        int n1 = DIM * CP;
        convert_bf16_kernel<<<(n1 + 255) / 256, 256, 0, stream>>>(w_dh, wdhB, n1);
        convert_bf16_kernel<<<(n1 + 255) / 256, 256, 0, stream>>>(w_dl, wdlB, n1);
        int n2 = CP * CP * 9;
        transpose_we1_kernel<<<(n2 + 255) / 256, 256, 0, stream>>>(w_e1, we1T);
    }
    // K1: 1x1 down-convs (WMMA)
    conv1x1_wmma_kernel<<<NB * HWH / 16, 128, 0, stream>>>(x_high, wdhB, b_dh, xhE, HWH);
    conv1x1_wmma_kernel<<<NB * HWL / 16, 128, 0, stream>>>(x_low,  wdlB, b_dl, xlE, HWL);
    // K2: upsample low embed, matcher conv + sigmoid
    {
        int tot = NB * DIM * HWH;
        resize_ac_kernel<<<(tot + 255) / 256, 256, 0, stream>>>(xlE, lowUp, NB * DIM,
                                                                HL, WL, HH, WH);
        int tp = NB * HWH;
        match_conv_kernel<<<(tp + 255) / 256, 256, 0, stream>>>(xhE, lowUp, w_match,
                                                                b_match, cert);
    }
    // K3: pools
    pool_kernel<<<1, 128, 0, stream>>>(cert, avgP, maxV, maxI);
    // K4: edge1 conv (WMMA + async LDS weight staging)
    conv3x3_edge1_wmma_kernel<<<NB * HWH / 16, 512, 0, stream>>>(
        x_high, we1T, avgP, bn_g, bn_b, bn_m, bn_v, hbuf);
    // K5: edge2 conv -> edge_pred + scratch
    {
        int tp = NB * HWH;
        edge2_conv_kernel<<<(tp + 255) / 256, 256, 0, stream>>>(hbuf, w_e2, outEdge, scratch);
    }
    // K6: top-64 points
    topk_kernel<<<NB, 256, 0, stream>>>(scratch, topkI, lowEI);
    // K7: edge-point gathers
    {
        int tp = NB * CP * PEDGE;
        gather_edge_kernel<<<(tp + 255) / 256, 256, 0, stream>>>(x_high, x_low, topkI,
                                                                 highE, lowE);
    }
    // K8: edge attention fusion
    attn_fuse_kernel<<<NB * PEDGE, 64, 0, stream>>>(highE, lowE, fusE);
    // K9: max-branch gathers
    {
        int tp = NB * CP * PEDGE;
        gather_max_kernel<<<(tp + 255) / 256, 256, 0, stream>>>(x_high, x_low, maxI,
                                                                maxV, highF, lowF, lowFI);
    }
    // K8': max-branch attention fusion
    attn_fuse_kernel<<<NB * PEDGE, 64, 0, stream>>>(highF, lowF, fusF);
    // K10: copy x_low into final output
    {
        int tp = NB * CP * HWL;
        copy_low_kernel<<<(tp + 255) / 256, 256, 0, stream>>>(x_low, outFinal);
    }
    // K11/K12: ordered scatters (edge first, then max branch overwrites)
    {
        int tp = NB * CP * PEDGE;
        scatter_kernel<<<(tp + 255) / 256, 256, 0, stream>>>(fusE, lowEI, outFinal);
        scatter_kernel<<<(tp + 255) / 256, 256, 0, stream>>>(fusF, lowFI, outFinal);
    }
}
